// TemporalSelfAttention_37374805410493
// MI455X (gfx1250) — matrix operand order repaired and checked by reading, not compiled, not run on hardware
//
#include <hip/hip_runtime.h>
#include <hip/hip_bf16.h>
#include <stdint.h>

// ---------------------------------------------------------------------------
// CDNA5 / gfx1250 wave32 WMMA types
// ---------------------------------------------------------------------------
typedef __attribute__((ext_vector_type(16))) __bf16 v16bf;
typedef __attribute__((ext_vector_type(8)))  float  v8f;

__device__ __forceinline__ uint32_t pack_bf16(float a, float b) {
  // round-to-nearest-even f32 -> bf16, packed pair (low = a, high = b)
  uint32_t ua = __float_as_uint(a);
  uint32_t ub = __float_as_uint(b);
  ua = (ua + 0x7FFFu + ((ua >> 16) & 1u)) >> 16;
  ub = (ub + 0x7FFFu + ((ub >> 16) & 1u)) >> 16;
  return (ua & 0xFFFFu) | (ub << 16);
}

// ---------------------------------------------------------------------------
// Tiled GEMM:  Y[M,N] = A[M,K0(+K1)] @ W[K,N] + bias (+ resid)
// A is a virtual concat: cols [0,K0) from A0; if SPLIT, cols [K0,K0+K1)
// from A1+A2 (elementwise sum).  bf16 WMMA 16x16x32, 64x64 block tile,
// 128 threads = 4 waves, each wave owns a 32x32 quadrant.
// K0, K1, M, N all multiples of the tile sizes for every launch below.
// ---------------------------------------------------------------------------
#define TILE_M 64
#define TILE_N 64
#define TILE_K 32
#define KP     16          // bf16 pairs per K tile
#define PITCH  20          // LDS row pitch in dwords: %4==0 -> b128 loads,
                           // stride 20 mod 64 banks -> conflict-free frags

// One K-tile: cooperative fill -> barrier -> fragments -> 4x WMMA -> barrier.
template<bool ADD2>
__device__ __forceinline__ void ktile(
    uint32_t* __restrict__ As, uint32_t* __restrict__ Bs,
    const float* __restrict__ A, const float* __restrict__ A2,
    int ldA, int ak,                       // k offset within this A segment
    const float* __restrict__ W, int wk,   // absolute k row into W
    int N, int mBase, int nBase, int tid, v8f (&acc)[2][2])
{
  const int lane = tid & 31;
  const int wave = tid >> 5;
  const int wm   = (wave >> 1) * 32;
  const int wn   = (wave & 1) * 32;
  const int half = lane >> 4;
  const int l15  = lane & 15;

  // ---- A tile: 64 rows x 32 k -> float4 per thread, 4 iterations ----------
#pragma unroll
  for (int it = 0; it < (TILE_M * (TILE_K / 4)) / 128; ++it) {   // 4 iters
    int idx = tid + it * 128;
    int r = idx >> 3, qd = idx & 7;        // row, quad-of-4-floats
    const float* ap = A + (size_t)(mBase + r) * ldA + (ak + 4 * qd);
    float4 f = *(const float4*)ap;
    if (ADD2) {
      float4 g = *(const float4*)(A2 + (size_t)(mBase + r) * ldA + (ak + 4 * qd));
      f.x += g.x; f.y += g.y; f.z += g.z; f.w += g.w;
    }
    __builtin_prefetch(ap + TILE_K, 0, 0);           // global_prefetch_b8
    As[r * PITCH + 2 * qd]     = pack_bf16(f.x, f.y);
    As[r * PITCH + 2 * qd + 1] = pack_bf16(f.z, f.w);
  }
  // ---- B tile, K-transposed: 64 cols x 16 pairs, 8 iterations -------------
#pragma unroll
  for (int it = 0; it < (TILE_N * KP) / 128; ++it) {
    int idx = tid + it * 128;
    int n = idx >> 4, p = idx & 15;
    int col = nBase + n;
    int k = wk + 2 * p;
    float f0 = W[(size_t)k * N + col];
    float f1 = W[(size_t)(k + 1) * N + col];
    Bs[n * PITCH + p] = pack_bf16(f0, f1);
  }
  __syncthreads();

  // ---- fragments per ISA 7.12.2 wave32 layouts (b128 LDS reads) -----------
  union BF { v16bf v; uint32_t u[8]; uint4 q[2]; };
  BF afrag[2], bfrag[2];
#pragma unroll
  for (int mt = 0; mt < 2; ++mt) {
    int r  = wm + mt * 16 + l15;           // A: lane&15 = row, K by lane half
    int kp = half * 4;
    afrag[mt].q[0] = *(const uint4*)&As[r * PITCH + kp];       // K 0..15 half
    afrag[mt].q[1] = *(const uint4*)&As[r * PITCH + 8 + kp];   // K 16..31 half
  }
#pragma unroll
  for (int nt = 0; nt < 2; ++nt) {
    int n  = wn + nt * 16 + l15;           // B: lane&15 = column
    int kp = half * 4;
    bfrag[nt].q[0] = *(const uint4*)&Bs[n * PITCH + kp];
    bfrag[nt].q[1] = *(const uint4*)&Bs[n * PITCH + kp + 8];
  }

  // ---- 4x v_wmma_f32_16x16x32_bf16 per wave per K step --------------------
#pragma unroll
  for (int mt = 0; mt < 2; ++mt)
#pragma unroll
    for (int nt = 0; nt < 2; ++nt)
      acc[mt][nt] = __builtin_amdgcn_wmma_f32_16x16x32_bf16(
          false, afrag[mt].v, false, bfrag[nt].v,
          (short)0, acc[mt][nt], false, false);

  __syncthreads();
}

template<bool SPLIT, bool HAS_RESID>
__global__ __launch_bounds__(128)
void gemm_bf16_wmma(const float* __restrict__ A0,
                    const float* __restrict__ A1,
                    const float* __restrict__ A2,
                    const float* __restrict__ W,
                    const float* __restrict__ bias,
                    const float* __restrict__ resid,
                    float* __restrict__ Y,
                    int N, int K0, int K1)
{
  __shared__ __align__(16) uint32_t As[TILE_M * PITCH];
  __shared__ __align__(16) uint32_t Bs[TILE_N * PITCH];

  const int tid   = threadIdx.x;
  const int lane  = tid & 31;
  const int wave  = tid >> 5;
  const int mBase = blockIdx.y * TILE_M;
  const int nBase = blockIdx.x * TILE_N;

  v8f acc[2][2];
#pragma unroll
  for (int i = 0; i < 2; ++i)
#pragma unroll
    for (int j = 0; j < 2; ++j)
      acc[i][j] = (v8f){0.f, 0.f, 0.f, 0.f, 0.f, 0.f, 0.f, 0.f};

  // segment 0: A0 columns [0, K0) — branch-free straight-line loop
  for (int kb = 0; kb < K0; kb += TILE_K)
    ktile<false>(As, Bs, A0, nullptr, K0, kb, W, kb, N, mBase, nBase, tid, acc);
  // segment 1 (qcat fusion): (A1 + A2) columns [K0, K0+K1)
  if (SPLIT)
    for (int kb = 0; kb < K1; kb += TILE_K)
      ktile<true>(As, Bs, A1, A2, K1, kb, W, K0 + kb, N, mBase, nBase, tid, acc);

  // ---- C/D layout: VGPR r -> row 8*(lane>>4)+r, col = lane&15 -------------
  const int wm   = (wave >> 1) * 32;
  const int wn   = (wave & 1) * 32;
  const int half = lane >> 4;
  const int l15  = lane & 15;
#pragma unroll
  for (int mt = 0; mt < 2; ++mt) {
#pragma unroll
    for (int nt = 0; nt < 2; ++nt) {
      int col  = nBase + wn + nt * 16 + l15;
      float bc = bias[col];
#pragma unroll
      for (int r = 0; r < 8; ++r) {
        int row = mBase + wm + mt * 16 + half * 8 + r;
        float val = acc[mt][nt][r] + bc;
        if (HAS_RESID) val += resid[(size_t)row * N + col];
        Y[(size_t)row * N + col] = val;
      }
    }
  }
}

// ---------------------------------------------------------------------------
// In-place softmax over groups of 4 contiguous logits (L*P = 4 per (q,h,b)).
// ---------------------------------------------------------------------------
__global__ __launch_bounds__(256)
void softmax4_inplace(float* __restrict__ a, int ngroups)
{
  int g = blockIdx.x * blockDim.x + threadIdx.x;
  if (g >= ngroups) return;
  float* p = a + (size_t)g * 4;
  float a0 = p[0], a1 = p[1], a2 = p[2], a3 = p[3];
  float m  = fmaxf(fmaxf(a0, a1), fmaxf(a2, a3));
  float e0 = __expf(a0 - m), e1 = __expf(a1 - m);
  float e2 = __expf(a2 - m), e3 = __expf(a3 - m);
  float inv = 1.0f / (e0 + e1 + e2 + e3);
  p[0] = e0 * inv; p[1] = e1 * inv; p[2] = e2 * inv; p[3] = e3 * inv;
}

// ---------------------------------------------------------------------------
// Deformable bilinear sampling + queue mean — branchless gather.
// One block per query, 8 waves = 8 heads, lane = channel (hd = 32 == wave32).
// v (82 MB) fits in MI455X's 192 MB L2, so all corner gathers are L2 hits;
// clamp+mask (as in the reference) lets all 4 corner loads per (b,p) issue
// unconditionally for maximum memory-level parallelism.
// ---------------------------------------------------------------------------
__global__ __launch_bounds__(256)
void deform_sample(const float* __restrict__ v,       // (2, 40000, 8, 32)
                   const float* __restrict__ off,     // (40000, 128)
                   const float* __restrict__ aw,      // (40000, 64)
                   const float* __restrict__ refpts,  // (2, 40000, 1, 2)
                   float* __restrict__ samp)          // (40000, 256)
{
  const int q = blockIdx.x;
  const int h = threadIdx.x >> 5;
  const int c = threadIdx.x & 31;
  const int Hh = 8, hd = 32, Hg = 200, Wg = 200, NQ = 40000;

  float acc = 0.0f;
#pragma unroll
  for (int b = 0; b < 2; ++b) {
    float rx = refpts[((size_t)b * NQ + q) * 2 + 0];
    float ry = refpts[((size_t)b * NQ + q) * 2 + 1];
#pragma unroll
    for (int p = 0; p < 4; ++p) {
      int   oi = (h * 2 + b) * 4 + p;
      float ox = off[(size_t)q * 128 + oi * 2 + 0];
      float oy = off[(size_t)q * 128 + oi * 2 + 1];
      float w  = aw [(size_t)q * 64  + oi];
      float x  = (rx + ox * (1.0f / Wg)) * Wg - 0.5f;
      float y  = (ry + oy * (1.0f / Hg)) * Hg - 0.5f;
      float x0f = floorf(x), y0f = floorf(y);
      float lx = x - x0f, ly = y - y0f;
      int   x0 = (int)x0f, y0 = (int)y0f;
      float s = 0.0f;
#pragma unroll
      for (int dy = 0; dy < 2; ++dy) {
#pragma unroll
        for (int dx = 0; dx < 2; ++dx) {
          int yi = y0 + dy, xi = x0 + dx;
          // branchless: clamp index, mask contribution
          int   yc = min(max(yi, 0), Hg - 1);
          int   xc = min(max(xi, 0), Wg - 1);
          float ok = (yi >= 0 && yi < Hg && xi >= 0 && xi < Wg) ? 1.0f : 0.0f;
          float wgt = (dy ? ly : 1.0f - ly) * (dx ? lx : 1.0f - lx);
          size_t vi = (((size_t)b * NQ + (size_t)yc * Wg + xc) * Hh + h) * hd + c;
          s = fmaf(wgt * ok, v[vi], s);
        }
      }
      acc = fmaf(w, s, acc);
    }
  }
  samp[(size_t)q * 256 + h * 32 + c] = acc * 0.5f;  // mean over 2 BEV queues
}

// ---------------------------------------------------------------------------
// Host-side orchestration
// ---------------------------------------------------------------------------
extern "C" void kernel_launch(void* const* d_in, const int* in_sizes, int n_in,
                              void* d_out, int out_size, void* d_ws, size_t ws_size,
                              hipStream_t stream)
{
  (void)in_sizes; (void)n_in; (void)out_size; (void)ws_size;

  const float* query  = (const float*)d_in[0];
  // d_in[1] key (dummy), d_in[5..7] shapes/indices/im2col unused (static here)
  const float* value  = (const float*)d_in[2];
  const float* qpos   = (const float*)d_in[3];
  const float* refpts = (const float*)d_in[4];
  const float* Wv     = (const float*)d_in[8];
  const float* bv     = (const float*)d_in[9];
  const float* Wo     = (const float*)d_in[10];
  const float* bo     = (const float*)d_in[11];
  const float* Wa     = (const float*)d_in[12];
  const float* ba     = (const float*)d_in[13];
  const float* Wout   = (const float*)d_in[14];
  const float* bout   = (const float*)d_in[15];
  float* out = (float*)d_out;

  const int NQ = 40000, C = 256;

  // workspace carve-up (floats)
  float* ws   = (float*)d_ws;
  float* v    = ws;                          // 80000*256
  float* offb = v    + (size_t)2 * NQ * C;   // 40000*128
  float* awb  = offb + (size_t)NQ * 128;     // 40000*64  (softmax in place)
  float* samp = awb  + (size_t)NQ * 64;      // 40000*256

  dim3 blk(128);

  // 1) v = value @ Wv + bv                 (M=80000, K=256, N=256)
  gemm_bf16_wmma<false, false><<<dim3(C / TILE_N, (2 * NQ) / TILE_M), blk, 0, stream>>>(
      value, nullptr, nullptr, Wv, bv, nullptr, v, C, C, 0);

  // 2) off = [value[:1] | query+qpos] @ Wo + bo   (M=40000, K=512, N=128)
  gemm_bf16_wmma<true, false><<<dim3(128 / TILE_N, NQ / TILE_M), blk, 0, stream>>>(
      value, query, qpos, Wo, bo, nullptr, offb, 128, C, C);

  // 3) aw logits = [value[:1] | query+qpos] @ Wa + ba   (M=40000, K=512, N=64)
  gemm_bf16_wmma<true, false><<<dim3(64 / TILE_N, NQ / TILE_M), blk, 0, stream>>>(
      value, query, qpos, Wa, ba, nullptr, awb, 64, C, C);

  // 4) softmax over the 4 points per (q, head, queue)
  {
    int ngroups = NQ * 16;                   // 8 heads * 2 queues
    softmax4_inplace<<<(ngroups + 255) / 256, 256, 0, stream>>>(awb, ngroups);
  }

  // 5) deformable gather + attention-weighted sum + queue mean
  deform_sample<<<NQ, 256, 0, stream>>>(v, offb, awb, refpts, samp);

  // 6) out = samp @ Wout + bout + query   (residual fused)
  gemm_bf16_wmma<false, true><<<dim3(C / TILE_N, NQ / TILE_M), blk, 0, stream>>>(
      samp, nullptr, nullptr, Wout, bout, query, out, C, C, 0);
}
